// Attention_6227702579820
// MI455X (gfx1250) — matrix operand order repaired and checked by reading, not compiled, not run on hardware
//
#include <hip/hip_runtime.h>
#include <hip/hip_bf16.h>

// ---------------- types ----------------
typedef _Float16 v16h __attribute__((ext_vector_type(16)));
typedef _Float16 v8h  __attribute__((ext_vector_type(8)));
typedef float    v8f  __attribute__((ext_vector_type(8)));

// ---------------- problem constants ----------------
#define DIM_   512
#define HEADS_ 8
#define DH_    64
#define INNER_ 512
#define NTOK_  256           // sequence length n
#define NBP_   256           // b*p = 4*64
#define ROWS_  65536         // b*p*n
#define NQKV_  1536          // 3*INNER
#define SCALE_ 0.125f        // 64^-0.5
#define LNEPS_ 1e-5f

// ---------------- WMMA helpers ----------------
__device__ __forceinline__ v8f wmma16(v16h a, v16h b, v8f c) {
  // D = A(16x32 f16) * B(32x16 f16) + C(16x16 f32)
  return __builtin_amdgcn_wmma_f32_16x16x32_f16(
      /*neg_a=*/false, a, /*neg_b=*/false, b,
      /*c_mod=*/(short)0, c, /*reuse_a=*/false, /*reuse_b=*/false);
}

// A-matrix 16x32 layout (contiguous k order, matches B): lane L holds row M=L&15;
// halves 0..15 = K(kbase+16*(L>=16) .. +15), kg = (L>=16?8:0)+kbase
__device__ __forceinline__ v16h load_afrag(const _Float16* p, int kg) {
  v8h lo = *(const v8h*)(p + kg + (kg & 8));
  v8h hi = *(const v8h*)(p + kg + (kg & 8) + 8);
  v16h r;
#pragma unroll
  for (int i = 0; i < 8; ++i) { r[i] = lo[i]; r[i + 8] = hi[i]; }
  return r;
}

// Stage a [64 x 512] f16 B panel (row-major, rows = output cols) into LDS with the
// gfx1250 async-copy path. 256 threads x 16 chunks x 16B = 64 KB. ASYNCcnt-tracked.
__device__ __forceinline__ void async_stage_b(_Float16* sB, const _Float16* gB,
                                              int tid) {
  uint32_t lds0 = (uint32_t)(uintptr_t)sB;  // low 32 bits = workgroup LDS offset
#pragma unroll
  for (int i = 0; i < 16; ++i) {
    int c = tid + i * 256;                  // 16-byte chunk index [0,4096)
    const _Float16* gp = gB + c * 8;
    uint32_t la = lds0 + c * 16;
    asm volatile("global_load_async_to_lds_b128 %0, %1, off"
                 :: "v"(la), "v"(gp) : "memory");
  }
  asm volatile("s_wait_asynccnt 0" ::: "memory");
  __syncthreads();  // all waves' DMA complete before any wave reads the panel
}

// ---------------- kernel 1: weight transpose + f16 convert ----------------
// src: fp32 [K, N] row-major (as stored for x @ W); dst: f16 [N, K] row-major
__global__ void wprep_kernel(const float* __restrict__ src, _Float16* __restrict__ dst,
                             int K, int N) {
  int i = blockIdx.x * blockDim.x + threadIdx.x;
  if (i >= K * N) return;
  int n = i / K;
  int k = i - n * K;
  dst[i] = (_Float16)src[(size_t)k * N + n];
}

// ---------------- kernel 2: LayerNorm (one wave per 512-elem row) ----------------
__global__ __launch_bounds__(256) void ln_kernel(const float* __restrict__ x,
                                                 const float* __restrict__ g,
                                                 const float* __restrict__ b,
                                                 _Float16* __restrict__ xn) {
  int row  = blockIdx.x * 8 + (threadIdx.x >> 5);
  int lane = threadIdx.x & 31;
  const float* xr = x + (size_t)row * DIM_;
  float4 v[4];
  float s = 0.f;
#pragma unroll
  for (int i = 0; i < 4; ++i) {
    v[i] = ((const float4*)xr)[lane * 4 + i];
    s += v[i].x + v[i].y + v[i].z + v[i].w;
  }
#pragma unroll
  for (int m = 16; m >= 1; m >>= 1) s += __shfl_xor(s, m, 32);
  float mu = s * (1.f / DIM_);
  float vs = 0.f;
#pragma unroll
  for (int i = 0; i < 4; ++i) {
    float dx0 = v[i].x - mu, dx1 = v[i].y - mu, dx2 = v[i].z - mu, dx3 = v[i].w - mu;
    vs += dx0 * dx0 + dx1 * dx1 + dx2 * dx2 + dx3 * dx3;
  }
#pragma unroll
  for (int m = 16; m >= 1; m >>= 1) vs += __shfl_xor(vs, m, 32);
  float r = rsqrtf(vs * (1.f / DIM_) + LNEPS_);
  _Float16* xo = xn + (size_t)row * DIM_;
#pragma unroll
  for (int i = 0; i < 4; ++i) {
    int base = lane * 16 + i * 4;
    float e0 = (v[i].x - mu) * r * g[base + 0] + b[base + 0];
    float e1 = (v[i].y - mu) * r * g[base + 1] + b[base + 1];
    float e2 = (v[i].z - mu) * r * g[base + 2] + b[base + 2];
    float e3 = (v[i].w - mu) * r * g[base + 3] + b[base + 3];
    xo[base + 0] = (_Float16)e0;
    xo[base + 1] = (_Float16)e1;
    xo[base + 2] = (_Float16)e2;
    xo[base + 3] = (_Float16)e3;
  }
}

// ---------------- kernel 3: QKV GEMM  [65536x512] x [512x1536] ----------------
// A = xn f16 row-major (global, prefetched); B panel staged to LDS via async copy.
// Writes Q,K row-major [bph, tok, d] and V transposed [bph, d, tok].
__global__ __launch_bounds__(256) void qkv_gemm_kernel(const _Float16* __restrict__ xn,
                                                       const _Float16* __restrict__ wT,
                                                       _Float16* __restrict__ q,
                                                       _Float16* __restrict__ k,
                                                       _Float16* __restrict__ vv) {
  __shared__ __attribute__((aligned(32))) _Float16 sB[64 * DIM_];  // 64 KB panel
  int wave = threadIdx.x >> 5;
  int lane = threadIdx.x & 31;
  int ln   = lane & 15;
  int hh   = lane >> 4;          // which half of the wave
  int m0   = blockIdx.x * 128 + wave * 16;
  int n0   = blockIdx.y * 64;

  async_stage_b(sB, wT + (size_t)n0 * DIM_, threadIdx.x);

  const _Float16* arow = xn + (size_t)(m0 + ln) * DIM_;
  v8f acc[4] = {};
#pragma unroll 4
  for (int kk = 0; kk < 16; ++kk) {
    int kg = kk * 32 + hh * 8;
    __builtin_prefetch(arow + kg + 128, 0, 1);   // global_prefetch_b8 on streamed A
    v16h a = load_afrag(arow, kg);
    int kb = kk * 32 + hh * 16;
#pragma unroll
    for (int s = 0; s < 4; ++s) {
      v16h bfrag = *(const v16h*)&sB[(s * 16 + ln) * DIM_ + kb];
      acc[s] = wmma16(a, bfrag, acc[s]);
    }
  }
  // scatter C fragments to Q/K row-major [bph, tok, d] and V transposed [bph, d, tok]
#pragma unroll
  for (int s = 0; s < 4; ++s) {
    int col = n0 + s * 16 + ln;       // [0,1536)
    int seg = col >> 9;               // 0=Q 1=K 2=V (uniform per subtile)
    int c   = col & 511;
    int h   = c >> 6;
    int d   = c & 63;
#pragma unroll
    for (int j = 0; j < 8; ++j) {
      int r   = m0 + j + hh * 8;
      int bp  = r >> 8;
      int tok = r & 255;
      size_t bph = (size_t)bp * HEADS_ + h;
      _Float16 val = (_Float16)acc[s][j];
      if (seg == 0)      q[(bph * NTOK_ + tok) * DH_ + d] = val;
      else if (seg == 1) k[(bph * NTOK_ + tok) * DH_ + d] = val;
      else               vv[(bph * DH_ + d) * NTOK_ + tok] = val;
    }
  }
}

// ---------------- kernel 4: attention per (b,p,h) x 64-query chunk ----------------
__global__ __launch_bounds__(128) void attn_kernel(const _Float16* __restrict__ q,
                                                   const _Float16* __restrict__ k,
                                                   const _Float16* __restrict__ vv,
                                                   _Float16* __restrict__ o) {
  __shared__ __attribute__((aligned(16))) _Float16 sp[4][16][NTOK_]; // probs, 32 KB
  int wave = threadIdx.x >> 5;
  int lane = threadIdx.x & 31;
  int ln   = lane & 15;
  int hh   = lane >> 4;
  int bph   = blockIdx.x >> 2;            // (b*p)*8 + h
  int chunk = blockIdx.x & 3;
  int qbase = chunk * 64 + wave * 16;

  const _Float16* qp = q + (size_t)bph * NTOK_ * DH_;
  const _Float16* kp = k + (size_t)bph * NTOK_ * DH_;
  const _Float16* vp = vv + (size_t)bph * DH_ * NTOK_;

  // Q A-fragments for this wave's 16 rows (contraction dim d = 64 -> 2 frags)
  const _Float16* qrow = qp + (size_t)(qbase + ln) * DH_;
  v16h qa0 = load_afrag(qrow, 0 + hh * 8);
  v16h qa1 = load_afrag(qrow, 32 + hh * 8);

  // scores: 16 column tiles of 16 keys
  v8f st[16];
#pragma unroll
  for (int t = 0; t < 16; ++t) {
    v8f c = {};
    const _Float16* krow = kp + (size_t)(t * 16 + ln) * DH_;
    v16h kb0 = *(const v16h*)(krow + hh * 16);
    v16h kb1 = *(const v16h*)(krow + 32 + hh * 16);
    c = wmma16(qa0, kb0, c);
    c = wmma16(qa1, kb1, c);
    st[t] = c;
  }

  // row-wise softmax directly on C-fragment layout: row M = j + 8*hh lives in one
  // 16-lane half; shfl_xor masks 1,2,4,8 reduce within that half only.
#pragma unroll
  for (int j = 0; j < 8; ++j) {
    float mx = -3.0e38f;
#pragma unroll
    for (int t = 0; t < 16; ++t) {
      float s = st[t][j] * SCALE_;
      st[t][j] = s;
      mx = fmaxf(mx, s);
    }
#pragma unroll
    for (int m = 8; m >= 1; m >>= 1) mx = fmaxf(mx, __shfl_xor(mx, m, 32));
    float sum = 0.f;
#pragma unroll
    for (int t = 0; t < 16; ++t) {
      float e = __expf(st[t][j] - mx);
      st[t][j] = e;
      sum += e;
    }
#pragma unroll
    for (int m = 8; m >= 1; m >>= 1) sum += __shfl_xor(sum, m, 32);
    float rinv = 1.f / sum;
    int row = j + hh * 8;
#pragma unroll
    for (int t = 0; t < 16; ++t)
      sp[wave][row][t * 16 + ln] = (_Float16)(st[t][j] * rinv);
  }
  __syncthreads();

  // out = probs[16x256] @ V[256x64]; B operand from V^T is contiguous in key dim
  v8f oa[4] = {};
  const _Float16* prow = &sp[wave][ln][0];
#pragma unroll
  for (int ks = 0; ks < 8; ++ks) {
    v16h a = load_afrag(prow, ks * 32 + hh * 8);
#pragma unroll
    for (int dt = 0; dt < 4; ++dt) {
      const _Float16* vrow = vp + (size_t)(dt * 16 + ln) * NTOK_ + ks * 32 + hh * 16;
      v16h bfrag = *(const v16h*)vrow;
      oa[dt] = wmma16(a, bfrag, oa[dt]);
    }
  }

  // write attn output in [b*p*n, INNER] row-major (head-interleaved) f16
  int h = bph & (HEADS_ - 1);
  size_t bp = (size_t)(bph >> 3);
#pragma unroll
  for (int dt = 0; dt < 4; ++dt) {
    int dim = dt * 16 + ln;
#pragma unroll
    for (int j = 0; j < 8; ++j) {
      int r = qbase + j + hh * 8;
      o[(bp * NTOK_ + r) * INNER_ + h * DH_ + dim] = (_Float16)oa[dt][j];
    }
  }
}

// ---------------- kernel 5: output projection + bias (fp32 out) ----------------
__global__ __launch_bounds__(256) void outproj_kernel(const _Float16* __restrict__ a,
                                                      const _Float16* __restrict__ wT,
                                                      const float* __restrict__ bias,
                                                      float* __restrict__ out) {
  __shared__ __attribute__((aligned(32))) _Float16 sB[64 * INNER_];  // 64 KB panel
  int wave = threadIdx.x >> 5;
  int lane = threadIdx.x & 31;
  int ln   = lane & 15;
  int hh   = lane >> 4;
  int m0   = blockIdx.x * 128 + wave * 16;
  int n0   = blockIdx.y * 64;

  async_stage_b(sB, wT + (size_t)n0 * INNER_, threadIdx.x);

  const _Float16* arow = a + (size_t)(m0 + ln) * INNER_;
  v8f acc[4] = {};
#pragma unroll 4
  for (int kk = 0; kk < 16; ++kk) {
    int kg = kk * 32 + hh * 8;
    __builtin_prefetch(arow + kg + 128, 0, 1);
    v16h af = load_afrag(arow, kg);
    int kb = kk * 32 + hh * 16;
#pragma unroll
    for (int s = 0; s < 4; ++s) {
      v16h bfrag = *(const v16h*)&sB[(s * 16 + ln) * INNER_ + kb];
      acc[s] = wmma16(af, bfrag, acc[s]);
    }
  }
#pragma unroll
  for (int s = 0; s < 4; ++s) {
    int col = n0 + s * 16 + ln;
    float bb = bias[col];
#pragma unroll
    for (int j = 0; j < 8; ++j) {
      int r = m0 + j + hh * 8;
      out[(size_t)r * DIM_ + col] = acc[s][j] + bb;
    }
  }
}

// ---------------- launch ----------------
extern "C" void kernel_launch(void* const* d_in, const int* in_sizes, int n_in,
                              void* d_out, int out_size, void* d_ws, size_t ws_size,
                              hipStream_t stream) {
  const float* x    = (const float*)d_in[0];
  const float* ln_g = (const float*)d_in[1];
  const float* ln_b = (const float*)d_in[2];
  const float* wqkv = (const float*)d_in[3];   // [512, 1536]
  const float* wout = (const float*)d_in[4];   // [512, 512]
  const float* bout = (const float*)d_in[5];
  float* out = (float*)d_out;

  char* ws = (char*)d_ws;
  size_t off = 0;
  _Float16* xn    = (_Float16*)(ws + off); off += (size_t)ROWS_ * DIM_ * 2;   // 64 MB
  _Float16* wqkvT = (_Float16*)(ws + off); off += (size_t)NQKV_ * DIM_ * 2;   // 1.5 MB
  _Float16* woutT = (_Float16*)(ws + off); off += (size_t)DIM_ * INNER_ * 2;  // 0.5 MB
  _Float16* qws   = (_Float16*)(ws + off); off += (size_t)ROWS_ * INNER_ * 2; // 64 MB
  _Float16* kws   = (_Float16*)(ws + off); off += (size_t)ROWS_ * INNER_ * 2; // 64 MB
  _Float16* vws   = (_Float16*)(ws + off); off += (size_t)ROWS_ * INNER_ * 2; // 64 MB
  _Float16* attno = xn;  // xn is fully consumed by QKV GEMM before attention writes

  wprep_kernel<<<(DIM_ * NQKV_ + 255) / 256, 256, 0, stream>>>(wqkv, wqkvT, DIM_, NQKV_);
  wprep_kernel<<<(INNER_ * DIM_ + 255) / 256, 256, 0, stream>>>(wout, woutT, INNER_, DIM_);
  ln_kernel<<<ROWS_ / 8, 256, 0, stream>>>(x, ln_g, ln_b, xn);
  qkv_gemm_kernel<<<dim3(ROWS_ / 128, NQKV_ / 64), 256, 0, stream>>>(xn, wqkvT, qws, kws, vws);
  attn_kernel<<<NBP_ * HEADS_ * 4, 128, 0, stream>>>(qws, kws, vws, attno);
  outproj_kernel<<<dim3(ROWS_ / 128, DIM_ / 64), 256, 0, stream>>>(attno, woutT, bout, out);
}
